// EnhancedDualRetriever_50096498541207
// MI455X (gfx1250) — matrix operand order repaired and checked by reading, not compiled, not run on hardware
//
#include <hip/hip_runtime.h>
#include <math.h>

typedef __attribute__((ext_vector_type(16))) _Float16 v16h;
typedef __attribute__((ext_vector_type(8)))  float    v8f;

#define TLEN   512
#define NCH    7
#define BATCH  64
#define DDIM   64
#define MKEYS  100000
#define SCALES 4
#define PRED   96
#define CHUNK  1024
#define SUB    128
#define NCHUNK 98            // ceil(100000/1024)
#define OUT0   (BATCH*PRED*NCH)   // 43008

// ---------------- workspace layout (bytes) ----------------
#define WS_XF    0                        // 64*512 f32        = 131072
#define WS_MODE  131072                   // 64 int            = 256
#define WS_QF16  131328                   // 4*64*64 f16       = 32768
#define WS_CANDV 164096                   // 4*64*98*5 f32     = 501760
#define WS_CANDI 665856                   // 4*64*98*5 int     = 501760
#define WS_RETR  1167616                  // 4*64*96 f32       = 98304
#define WS_TOP1  1265920                  // 256 f32
#define WS_PVDR  1266944                  // 256 f32

__device__ __forceinline__ float log_marg(float n, float mean, float var,
                                          float pm, float pv, float nv) {
  float post_var  = 1.f / (1.f / pv + n / nv);
  float post_mean = post_var * (pm / pv + n * mean / nv);
  return -n * 0.5f * logf(2.f * 3.14159265358979f * nv)
       + 0.5f * logf(post_var / pv)
       - 0.5f * (n * var / nv + mean * mean * n / nv
                 - post_mean * post_mean / post_var + pm * pm / pv);
}

// branchless sorted insert of (v, m) into descending top-5
__device__ __forceinline__ void top5_insert(float (&tv)[5], int (&ti)[5],
                                            float v, int m) {
  if (v > tv[4]) {
    float nv2 = v; int ni = m;
    #pragma unroll
    for (int j = 0; j < 5; ++j) {
      bool gt = nv2 > tv[j];
      float cv = gt ? nv2 : tv[j]; int ci = gt ? ni : ti[j];
      nv2 = gt ? tv[j] : nv2;      ni = gt ? ti[j] : ni;
      tv[j] = cv; ti[j] = ci;
    }
  }
}

// ---------------- kernel 1: stats + changepoint + mode ----------------
__global__ void __launch_bounds__(512)
k1_stats_cp(const float* __restrict__ x,
            const float* __restrict__ cls_w, const float* __restrict__ cls_b,
            const float* __restrict__ prior_mean, const float* __restrict__ prior_var,
            const float* __restrict__ noise_var,
            float* __restrict__ xf_ws, int* __restrict__ mode_ws) {
  const int b = blockIdx.x;
  const int tid = threadIdx.x;                 // 512 threads = 16 waves
  __shared__ float c1[TLEN];
  __shared__ float c2[TLEN];
  __shared__ float red[TLEN];
  __shared__ float chanSum[NCH], chanSq[NCH], chanMax[NCH], chanMin[NCH];
  __shared__ float sh_extreme;

  const float* xb = x + (size_t)b * TLEN * NCH;

  // xf[t] = mean over channels
  float v = 0.f;
  #pragma unroll
  for (int c = 0; c < NCH; ++c) v += xb[tid * NCH + c];
  v *= (1.f / NCH);
  xf_ws[b * TLEN + tid] = v;

  // per-channel stats: waves 0..6 each take one channel
  const int wave = tid >> 5, lane = tid & 31;
  if (wave < NCH) {
    const int c = wave;
    float s = 0.f, sq = 0.f, mx = -INFINITY, mn = INFINITY;
    for (int t = lane; t < TLEN; t += 32) {
      float u = xb[t * NCH + c];
      s += u; sq += u * u; mx = fmaxf(mx, u); mn = fminf(mn, u);
    }
    #pragma unroll
    for (int off = 16; off > 0; off >>= 1) {
      s  += __shfl_xor(s,  off, 32);
      sq += __shfl_xor(sq, off, 32);
      mx  = fmaxf(mx, __shfl_xor(mx, off, 32));
      mn  = fminf(mn, __shfl_xor(mn, off, 32));
    }
    if (lane == 0) { chanSum[c] = s; chanSq[c] = sq; chanMax[c] = mx; chanMin[c] = mn; }
  }
  __syncthreads();
  if (tid == 0) {
    float f0 = 0, f1 = 0, f2 = 0, f3 = 0, f4 = 0;
    for (int c = 0; c < NCH; ++c) {
      float m = chanSum[c] / (float)TLEN;
      float var = (chanSq[c] - TLEN * m * m) / (float)(TLEN - 1);
      f0 += m;
      f1 += sqrtf(fmaxf(var, 0.f));
      f2 += chanMax[c];
      f3 += chanMin[c];
      f4 += xb[(TLEN - 1) * NCH + c] - xb[c];
    }
    f0 /= NCH; f1 /= NCH; f2 /= NCH; f3 /= NCH; f4 /= NCH;
    f1 = fmaxf(f1, 1e-6f);
    float z = f0 * cls_w[0] + f1 * cls_w[1] + f2 * cls_w[2]
            + f3 * cls_w[3] + f4 * cls_w[4] + cls_b[0];
    sh_extreme = 1.f / (1.f + expf(-z));
  }

  // inclusive scan of xf and xf^2 (Hillis-Steele)
  c1[tid] = v; c2[tid] = v * v;
  __syncthreads();
  for (int off = 1; off < TLEN; off <<= 1) {
    float a  = (tid >= off) ? c1[tid - off] : 0.f;
    float bb = (tid >= off) ? c2[tid - off] : 0.f;
    __syncthreads();
    c1[tid] += a; c2[tid] += bb;
    __syncthreads();
  }
  const float s1T = c1[TLEN - 1], s2T = c2[TLEN - 1];
  const float pm = prior_mean[0];
  const float pv = logf(1.f + expf(prior_var[0]));
  const float nv = logf(1.f + expf(noise_var[0]));

  float bf = -INFINITY;
  const bool valid = (tid >= 16) && (tid < TLEN - 16);
  if (valid) {
    const int pos = tid;
    float nl = (float)pos, nr = (float)(TLEN - pos);
    float s1p = c1[pos - 1], s2p = c2[pos - 1];
    float ml = s1p / nl;
    float vl = fmaxf((s2p - nl * ml * ml) / fmaxf(nl - 1.f, 1.f), 1e-8f);
    float mr = (s1T - s1p) / nr;
    float vr = fmaxf(((s2T - s2p) - nr * mr * mr) / fmaxf(nr - 1.f, 1.f), 1e-8f);
    float mw = s1T / (float)TLEN;
    float vw = fmaxf((s2T - TLEN * mw * mw) / (float)(TLEN - 1), 1e-8f);
    bf = log_marg(nl, ml, vl, pm, pv, nv)
       + log_marg(nr, mr, vr, pm, pv, nv)
       - log_marg((float)TLEN, mw, vw, pm, pv, nv);
  }

  // block max
  red[tid] = bf;
  __syncthreads();
  for (int off = 256; off > 0; off >>= 1) {
    if (tid < off) red[tid] = fmaxf(red[tid], red[tid + off]);
    __syncthreads();
  }
  const float maxbf = red[0];
  __syncthreads();
  // total softmax denominator
  const float e = valid ? expf(bf - maxbf) : 0.f;
  red[tid] = e;
  __syncthreads();
  for (int off = 256; off > 0; off >>= 1) {
    if (tid < off) red[tid] += red[tid + off];
    __syncthreads();
  }
  const float tot = red[0];
  __syncthreads();
  // masked sum: pos > int(512*0.8) = 409
  red[tid] = (valid && tid > 409) ? e : 0.f;
  __syncthreads();
  for (int off = 256; off > 0; off >>= 1) {
    if (tid < off) red[tid] += red[tid + off];
    __syncthreads();
  }
  if (tid == 0) {
    float near_end = (1.f / (1.f + expf(-maxbf))) * (red[0] / tot);
    mode_ws[b] = (near_end > 0.5f) ? 2 : ((sh_extreme > 0.5f) ? 1 : 0);
  }
}

// ---------------- kernel 2: per-scale query encoder -> f16 ----------------
__global__ void __launch_bounds__(64)
k2_encode(const float* __restrict__ xf_ws,
          const float* __restrict__ enc_W, const float* __restrict__ enc_b,
          const float* __restrict__ ln_g, const float* __restrict__ ln_b,
          _Float16* __restrict__ q_ws) {
  const int s = blockIdx.x >> 6;
  const int b = blockIdx.x & 63;
  const int tid = threadIdx.x;                  // 64 threads
  const int ds = 1 << s;
  const int len = TLEN >> s;
  const float* xr = xf_ws + b * TLEN;
  __shared__ float r[64];
  __shared__ float stats[5];

  auto redsum = [&](float val) -> float {
    r[tid] = val; __syncthreads();
    for (int off = 32; off > 0; off >>= 1) { if (tid < off) r[tid] += r[tid + off]; __syncthreads(); }
    float out = r[0]; __syncthreads(); return out;
  };
  auto redmax = [&](float val) -> float {
    r[tid] = val; __syncthreads();
    for (int off = 32; off > 0; off >>= 1) { if (tid < off) r[tid] = fmaxf(r[tid], r[tid + off]); __syncthreads(); }
    float out = r[0]; __syncthreads(); return out;
  };
  auto redmin = [&](float val) -> float {
    r[tid] = val; __syncthreads();
    for (int off = 32; off > 0; off >>= 1) { if (tid < off) r[tid] = fminf(r[tid], r[tid + off]); __syncthreads(); }
    float out = r[0]; __syncthreads(); return out;
  };

  float sum = 0.f, sq = 0.f, mx = -INFINITY, mn = INFINITY;
  for (int i = tid; i < len; i += 64) {
    float u = 0.f;
    for (int j = 0; j < ds; ++j) u += xr[i * ds + j];
    u /= (float)ds;
    sum += u; sq += u * u; mx = fmaxf(mx, u); mn = fminf(mn, u);
  }
  float tsum = redsum(sum);
  float tsq  = redsum(sq);
  float tmx  = redmax(mx);
  float tmn  = redmin(mn);
  if (tid == 0) {
    float mean = tsum / (float)len;
    float var  = (tsq - len * mean * mean) / (float)(len - 1);
    float sd   = fmaxf(sqrtf(fmaxf(var, 0.f)), 1e-6f);
    float fst = 0.f, lst = 0.f;
    for (int j = 0; j < ds; ++j) { fst += xr[j]; lst += xr[(len - 1) * ds + j]; }
    fst /= (float)ds; lst /= (float)ds;
    stats[0] = mean; stats[1] = sd; stats[2] = tmx; stats[3] = tmn; stats[4] = lst - fst;
  }
  __syncthreads();

  float h = enc_b[tid];
  #pragma unroll
  for (int j = 0; j < 5; ++j) h += stats[j] * enc_W[j * DDIM + tid];
  float hs = redsum(h);
  float hq = redsum(h * h);
  float mu = hs / 64.f;
  float varh = hq / 64.f - mu * mu;
  float g = (h - mu) * rsqrtf(varh + 1e-5f) * ln_g[tid] + ln_b[tid];
  float nn = redsum(g * g);
  float qv = g / sqrtf(nn);
  q_ws[(size_t)(s * 64 + b) * DDIM + tid] = (_Float16)qv;
}

// ---------------- kernel 3: WMMA retrieval + chunk top-5 ----------------
__global__ void __launch_bounds__(256)
k3_retrieve(const float* __restrict__ keys,
            const int* __restrict__ labels,
            const _Float16* __restrict__ q_ws,
            const int* __restrict__ mode_ws,
            float* __restrict__ cand_v, int* __restrict__ cand_i) {
  const int s = blockIdx.y;
  const int base = blockIdx.x * CHUNK;
  const int tid = threadIdx.x;
  const int wave = tid >> 5, lane = tid & 31;
  const int qt = wave & 3;          // query tile 0..3 (rows of 16)
  const int kg = wave >> 2;         // key-tile group 0..1
  const int lrow = lane & 15, g = lane >> 4;

  __shared__ float sims[64][SUB + 1];   // +1 pad: scan banks = (q + k) % 64
  __shared__ int   lab[SUB];
  __shared__ float mv[64][21];          // 4x top-5 per query (padded rows)
  __shared__ int   mi[64][21];

  // A tile (16x32 f16 layout): lane group g holds 16 contiguous K in [16g,16g+16)
  const _Float16* qrow = q_ws + (size_t)(s * 64 + qt * 16 + lrow) * DDIM;
  v16h a0, a1;
  #pragma unroll
  for (int i = 0; i < 8; ++i) {
    a0[i]     = qrow[g * 16 + i];
    a0[8 + i] = qrow[g * 16 + 8 + i];
    a1[i]     = qrow[32 + g * 16 + i];
    a1[8 + i] = qrow[32 + g * 16 + 8 + i];
  }

  // scanner mapping: 4 threads per query; thread scans 32-key slice per sub-chunk
  const int q  = tid & 63;
  const int qd = tid >> 6;               // 0..3
  const int mode = mode_ws[q];

  float tv[5]; int ti[5];
  #pragma unroll
  for (int i = 0; i < 5; ++i) { tv[i] = -INFINITY; ti[i] = 0; }

  for (int sc = 0; sc < CHUNK / SUB; ++sc) {
    const int sbase = base + sc * SUB;
    if (tid < SUB) {
      int m = sbase + tid;
      lab[tid] = (m < MKEYS) ? labels[(size_t)s * MKEYS + m] : -1;
    }
    // prefetch next sub-chunk's key rows
    {
      int mp = sbase + SUB + (kg * 4) * 16 + lrow;
      if (mp < MKEYS)
        __builtin_prefetch(keys + ((size_t)s * MKEYS + mp) * DDIM, 0, 1);
    }
    #pragma unroll
    for (int t = 0; t < 4; ++t) {
      const int ktLocal = kg * 4 + t;
      int m = sbase + ktLocal * 16 + lrow;
      if (m >= MKEYS) m = MKEYS - 1;
      const float* kr = keys + ((size_t)s * MKEYS + m) * DDIM;
      // B tile (32x16 f16): lane group g holds 16 contiguous K values
      v16h b0, b1;
      #pragma unroll
      for (int i = 0; i < 16; ++i) b0[i] = (_Float16)kr[g * 16 + i];
      #pragma unroll
      for (int i = 0; i < 16; ++i) b1[i] = (_Float16)kr[32 + g * 16 + i];
      v8f c = {};
      c = __builtin_amdgcn_wmma_f32_16x16x32_f16(false, a0, false, b0, (short)0, c, false, false);
      c = __builtin_amdgcn_wmma_f32_16x16x32_f16(false, a1, false, b1, (short)0, c, false, false);
      const int mrow = qt * 16 + ((lane & 16) ? 8 : 0);
      const int col = ktLocal * 16 + lrow;
      #pragma unroll
      for (int rr = 0; rr < 8; ++rr) sims[mrow + rr][col] = c[rr];
    }
    __syncthreads();
    // per-query partial top-5: all 256 threads scan (4 per query, 32 keys each)
    {
      const int kbase = qd * 32;
      for (int j = 0; j < 32; ++j) {
        int k = kbase + j;
        int m = sbase + k;
        bool allowed = ((mode == 0) || (lab[k] == mode)) && (m < MKEYS);
        float v = allowed ? sims[q][k] : -INFINITY;
        top5_insert(tv, ti, v, m);
      }
    }
    __syncthreads();
  }

  // stage partials and merge 4x5 -> 5 per query
  #pragma unroll
  for (int i = 0; i < 5; ++i) { mv[q][qd * 5 + i] = tv[i]; mi[q][qd * 5 + i] = ti[i]; }
  __syncthreads();
  if (tid < 64) {
    float fv[5]; int fi[5];
    #pragma unroll
    for (int i = 0; i < 5; ++i) { fv[i] = -INFINITY; fi[i] = 0; }
    for (int k = 0; k < 20; ++k) {
      top5_insert(fv, fi, mv[tid][k], mi[tid][k]);
    }
    size_t o = ((size_t)((s * 64 + tid) * NCHUNK) + blockIdx.x) * 5;
    #pragma unroll
    for (int i = 0; i < 5; ++i) { cand_v[o + i] = fv[i]; cand_i[o + i] = fi[i]; }
  }
}

// ---------------- kernel 4: global top-5 + retrieve values ----------------
__global__ void __launch_bounds__(128)
k4_reduce(const float* __restrict__ cand_v, const int* __restrict__ cand_i,
          const float* __restrict__ values, const float* __restrict__ thresholds,
          float* __restrict__ retr_ws, float* __restrict__ top1_ws,
          float* __restrict__ pvdr_ws) {
  const int s = blockIdx.x >> 6, b = blockIdx.x & 63;
  const int tid = threadIdx.x;
  __shared__ float w[5];
  __shared__ int idx[5];
  if (tid == 0) {
    float tv[5]; int ti[5];
    #pragma unroll
    for (int i = 0; i < 5; ++i) { tv[i] = -INFINITY; ti[i] = 0; }
    size_t o = (size_t)(s * 64 + b) * NCHUNK * 5;
    for (int k = 0; k < NCHUNK * 5; ++k) {
      top5_insert(tv, ti, cand_v[o + k], cand_i[o + k]);
    }
    float m0 = tv[0];
    float e[5], tot = 0.f;
    #pragma unroll
    for (int i = 0; i < 5; ++i) { e[i] = expf(tv[i] - m0); tot += e[i]; }
    #pragma unroll
    for (int i = 0; i < 5; ++i) { w[i] = e[i] / tot; idx[i] = ti[i]; }
    top1_ws[s * 64 + b] = m0;
    pvdr_ws[s * 64 + b] = 1.f / (1.f + expf(-(m0 - thresholds[s])));
  }
  __syncthreads();
  if (tid < PRED) {
    float acc = 0.f;
    #pragma unroll
    for (int i = 0; i < 5; ++i)
      acc += w[i] * values[((size_t)s * MKEYS + idx[i]) * PRED + tid];
    retr_ws[(size_t)(s * 64 + b) * PRED + tid] = acc;
  }
}

// ---------------- kernel 5: cross-scale fuse + output ----------------
__global__ void __launch_bounds__(128)
k5_fuse(const float* __restrict__ retr_ws, const float* __restrict__ top1_ws,
        const float* __restrict__ pvdr_ws, float* __restrict__ out) {
  const int b = blockIdx.x;
  const int tid = threadIdx.x;
  __shared__ float sw[SCALES];
  if (tid == 0) {
    float t0 = top1_ws[b], t1 = top1_ws[64 + b], t2 = top1_ws[128 + b], t3 = top1_ws[192 + b];
    float m = fmaxf(fmaxf(t0, t1), fmaxf(t2, t3));
    float e0 = expf(t0 - m), e1 = expf(t1 - m), e2 = expf(t2 - m), e3 = expf(t3 - m);
    float tot = e0 + e1 + e2 + e3;
    sw[0] = e0 / tot; sw[1] = e1 / tot; sw[2] = e2 / tot; sw[3] = e3 / tot;
  }
  __syncthreads();
  if (tid < PRED) {
    float f = 0.f;
    #pragma unroll
    for (int s = 0; s < SCALES; ++s) f += sw[s] * retr_ws[(size_t)(s * 64 + b) * PRED + tid];
    float* ob = out + (size_t)b * PRED * NCH + tid * NCH;
    #pragma unroll
    for (int n = 0; n < NCH; ++n) ob[n] = f;
  }
  if (tid < SCALES) out[OUT0 + b * SCALES + tid] = pvdr_ws[tid * 64 + b];
}

// ---------------- launch ----------------
extern "C" void kernel_launch(void* const* d_in, const int* in_sizes, int n_in,
                              void* d_out, int out_size, void* d_ws, size_t ws_size,
                              hipStream_t stream) {
  (void)in_sizes; (void)n_in; (void)out_size; (void)ws_size;
  const float* x          = (const float*)d_in[0];
  const float* keys       = (const float*)d_in[1];
  const float* values     = (const float*)d_in[2];
  const int*   labels     = (const int*)  d_in[3];
  const float* thresholds = (const float*)d_in[4];
  const float* cls_w      = (const float*)d_in[5];
  const float* cls_b      = (const float*)d_in[6];
  const float* prior_mean = (const float*)d_in[7];
  const float* prior_var  = (const float*)d_in[8];
  const float* noise_var  = (const float*)d_in[9];
  const float* enc_W      = (const float*)d_in[10];
  const float* enc_b      = (const float*)d_in[11];
  const float* ln_g       = (const float*)d_in[12];
  const float* ln_b       = (const float*)d_in[13];

  char* ws = (char*)d_ws;
  float*     xf_ws   = (float*)    (ws + WS_XF);
  int*       mode_ws = (int*)      (ws + WS_MODE);
  _Float16*  q_ws    = (_Float16*) (ws + WS_QF16);
  float*     cand_v  = (float*)    (ws + WS_CANDV);
  int*       cand_i  = (int*)      (ws + WS_CANDI);
  float*     retr_ws = (float*)    (ws + WS_RETR);
  float*     top1_ws = (float*)    (ws + WS_TOP1);
  float*     pvdr_ws = (float*)    (ws + WS_PVDR);

  k1_stats_cp<<<BATCH, 512, 0, stream>>>(x, cls_w, cls_b, prior_mean, prior_var,
                                         noise_var, xf_ws, mode_ws);
  k2_encode<<<SCALES * BATCH, 64, 0, stream>>>(xf_ws, enc_W, enc_b, ln_g, ln_b, q_ws);
  dim3 g3(NCHUNK, SCALES);
  k3_retrieve<<<g3, 256, 0, stream>>>(keys, labels, q_ws, mode_ws, cand_v, cand_i);
  k4_reduce<<<SCALES * BATCH, 128, 0, stream>>>(cand_v, cand_i, values, thresholds,
                                                retr_ws, top1_ws, pvdr_ws);
  k5_fuse<<<BATCH, 128, 0, stream>>>(retr_ws, top1_ws, pvdr_ws, (float*)d_out);
}
